// SingleStreamBlock_41274635714822
// MI455X (gfx1250) — matrix operand order repaired and checked by reading, not compiled, not run on hardware
//
#include <hip/hip_runtime.h>
#include <hip/hip_bf16.h>
#include <math.h>

typedef __attribute__((ext_vector_type(16))) __bf16 v16bf;
typedef __attribute__((ext_vector_type(8)))  __bf16 v8bf;
typedef __attribute__((ext_vector_type(8)))  float  v8f;

// Problem dims
constexpr int Lh   = 2048;
constexpr int Hh   = 3072;
constexpr int NHh  = 24;
constexpr int Dh   = 128;
constexpr int MLPh = 12288;
constexpr int N1h  = 3 * Hh + MLPh;   // 21504
constexpr int K2h  = Hh + MLPh;       // 15360

#define WMMA_BF16(a, b, c) \
  __builtin_amdgcn_wmma_f32_16x16x32_bf16(false, (a), false, (b), (short)0, (c), false, false)
#define CAT16(lo, hi) \
  __builtin_shufflevector((lo), (hi), 0, 1, 2, 3, 4, 5, 6, 7, 8, 9, 10, 11, 12, 13, 14, 15)

// ---------------------------------------------------------------- tiled transpose + cast: w[K][N] f32 -> wt[N][K] bf16
__global__ void k_castT(const float* __restrict__ w, __bf16* __restrict__ wt, int K, int N) {
  __shared__ float tile[32][33];
  int k0 = blockIdx.x * 32, n0 = blockIdx.y * 32;
  int tx = threadIdx.x & 31, ty = threadIdx.x >> 5;  // ty: 0..7
#pragma unroll
  for (int p = 0; p < 4; ++p)
    tile[ty + p * 8][tx] = w[(size_t)(k0 + ty + p * 8) * N + n0 + tx];
  __syncthreads();
#pragma unroll
  for (int p = 0; p < 4; ++p)
    wt[(size_t)(n0 + ty + p * 8) * K + k0 + tx] = (__bf16)tile[tx][ty + p * 8];
}

// ---------------------------------------------------------------- mod = silu(vec)@mod_w + mod_b
__global__ void k_mod(const float* __restrict__ vec, const float* __restrict__ mw,
                      const float* __restrict__ mb, float* __restrict__ mod) {
  int j = blockIdx.x * 256 + threadIdx.x;
  if (j >= 3 * Hh) return;
  float s = 0.f;
  for (int i = 0; i < Hh; ++i) {
    float v  = vec[i];
    float sv = v / (1.f + __expf(-v));
    s += sv * mw[(size_t)i * (3 * Hh) + j];
  }
  mod[j] = s + mb[j];
}

// ---------------------------------------------------------------- LayerNorm + modulate -> bf16
__global__ void k_ln_mod(const float* __restrict__ x, const float* __restrict__ mod,
                         const float* __restrict__ lns, const float* __restrict__ lnb,
                         __bf16* __restrict__ xmod) {
  __shared__ float red[256];
  int row = blockIdx.x, tid = threadIdx.x;
  const float* xr = x + (size_t)row * Hh;
  float s = 0.f;
  for (int i = tid; i < Hh; i += 256) s += xr[i];
  red[tid] = s; __syncthreads();
  for (int st = 128; st > 0; st >>= 1) { if (tid < st) red[tid] += red[tid + st]; __syncthreads(); }
  float mu = red[0] * (1.f / Hh);
  __syncthreads();
  float vs = 0.f;
  for (int i = tid; i < Hh; i += 256) { float d = xr[i] - mu; vs += d * d; }
  red[tid] = vs; __syncthreads();
  for (int st = 128; st > 0; st >>= 1) { if (tid < st) red[tid] += red[tid + st]; __syncthreads(); }
  float rstd = rsqrtf(red[0] * (1.f / Hh) + 1e-6f);
  for (int i = tid; i < Hh; i += 256) {
    float xn = (xr[i] - mu) * rstd * lns[i] + lnb[i];
    xmod[(size_t)row * Hh + i] = (__bf16)((1.f + mod[Hh + i]) * xn + mod[i]);
  }
}

// ---------------------------------------------------------------- WMMA GEMM
// C(MxN) = A(MxK bf16, row-major) @ Bt(NxK bf16, i.e. B transposed) + bias.
// 256 thr / 8 waves; block tile 128x128, K-step 32; wave tile 32x64 (2x4 frags).
// A tile staged via global_load_async_to_lds_b128 (K-chunk permuted, fragment-contiguous),
// B tile staged regs->LDS in fragment-major layout; double-buffered software pipeline.
__global__ __launch_bounds__(256) void k_gemm(
    const __bf16* __restrict__ A, const __bf16* __restrict__ Bt,
    const float* __restrict__ bias, float* __restrict__ C,
    int M, int N, int K) {
  __shared__ __bf16 As[2][128 * 40];
  __shared__ __bf16 Bs[2][8 * 2 * 16 * 24];  // [frag][half][lp][24]
  int tid = threadIdx.x;
  int lane = tid & 31, wid = tid >> 5;
  int half = lane >> 4, lp = lane & 15;
  int wm = wid & 3, wn = wid >> 2;
  int m0 = blockIdx.y * 128, n0 = blockIdx.x * 128;

  // staging geometry (per thread)
  int ar  = tid >> 1;                    // A row 0..127
  int ac8 = (tid & 1) * 2;               // A chunk pair (8-half chunks)
  int pcA0 = ((ac8 & 1) << 1) | (ac8 >> 1);
  int pcA1 = (((ac8 + 1) & 1) << 1) | ((ac8 + 1) >> 1);
  int col = tid >> 1, ch = tid & 1;      // B col 0..127, K-half 0..1
  int bo  = (((col >> 4) * 2 + ch) * 16 + (col & 15)) * 24;

  v8f acc[2][4] = {};
  int nt = K / 32;
  for (int it = 0; it <= nt; ++it) {
    if (it < nt) {
      int k0  = it * 32;
      int buf = it & 1;
      // --- async DMA: A tile -> LDS (fragment-permuted chunks)
      unsigned l0 = (unsigned)(size_t)&As[buf][ar * 40 + pcA0 * 8];
      unsigned l1 = (unsigned)(size_t)&As[buf][ar * 40 + pcA1 * 8];
      unsigned long long g0 = (unsigned long long)(size_t)&A[(size_t)(m0 + ar) * K + k0 + ac8 * 8];
      unsigned long long g1 = g0 + 16;
      asm volatile("global_load_async_to_lds_b128 %0, %1, off" :: "v"(l0), "v"(g0) : "memory");
      asm volatile("global_load_async_to_lds_b128 %0, %1, off" :: "v"(l1), "v"(g1) : "memory");
      // --- B tile: K-contiguous from transposed weights, stored fragment-major
      const __bf16* gb = &Bt[(size_t)(n0 + col) * K + k0 + ch * 8];
      v8bf b0 = *(const v8bf*)gb;
      v8bf b1 = *(const v8bf*)(gb + 16);
      *(v8bf*)&Bs[buf][bo]     = b0;
      *(v8bf*)&Bs[buf][bo + 8] = b1;
    }
    if (it > 0) {
      if (it < nt) asm volatile("s_wait_asynccnt 0x2" ::: "memory");
      else         asm volatile("s_wait_asynccnt 0x0" ::: "memory");
      __syncthreads();
      int buf = (it - 1) & 1;
      v16bf a[2], b[4];
#pragma unroll
      for (int fm = 0; fm < 2; ++fm) {
        int mr = wm * 32 + fm * 16 + lp;
        v8bf lo = *(const v8bf*)&As[buf][mr * 40 + half * 16];
        v8bf hi = *(const v8bf*)&As[buf][mr * 40 + half * 16 + 8];
        a[fm] = CAT16(lo, hi);
      }
#pragma unroll
      for (int fn = 0; fn < 4; ++fn) {
        int frag = wn * 4 + fn;
        const __bf16* bp = &Bs[buf][((frag * 2 + half) * 16 + lp) * 24];
        v8bf lo = *(const v8bf*)bp;
        v8bf hi = *(const v8bf*)(bp + 8);
        b[fn] = CAT16(lo, hi);
      }
#pragma unroll
      for (int fm = 0; fm < 2; ++fm)
#pragma unroll
        for (int fn = 0; fn < 4; ++fn)
          acc[fm][fn] = WMMA_BF16(a[fm], b[fn], acc[fm][fn]);
      __syncthreads();
    }
  }
  for (int fm = 0; fm < 2; ++fm)
    for (int fn = 0; fn < 4; ++fn) {
      int col_ = n0 + wn * 64 + fn * 16 + lp;
      float bv = bias[col_];
      for (int i = 0; i < 8; ++i) {
        int row = m0 + wm * 32 + fm * 16 + half * 8 + i;
        C[(size_t)row * N + col_] = acc[fm][fn][i] + bv;
      }
    }
}

// ---------------------------------------------------------------- RMSNorm(q,k)+RoPE -> bf16 q,k,v in [NH][L][D]
__global__ void k_qkrope(const float* __restrict__ h, const float* __restrict__ pe,
                         const float* __restrict__ qs, const float* __restrict__ ks,
                         __bf16* __restrict__ qb, __bf16* __restrict__ kb, __bf16* __restrict__ vb) {
  __shared__ float qsh[128], ksh[128], red[128];
  int b = blockIdx.x;
  int t = b / NHh, hh = b % NHh;
  int d = threadIdx.x;
  const float* hr = h + (size_t)t * N1h;
  float qv = hr[hh * Dh + d];
  float kv = hr[Hh + hh * Dh + d];
  float vv = hr[2 * Hh + hh * Dh + d];
  red[d] = qv * qv; __syncthreads();
  for (int st = 64; st > 0; st >>= 1) { if (d < st) red[d] += red[d + st]; __syncthreads(); }
  float qr = rsqrtf(red[0] * (1.f / Dh) + 1e-6f);
  __syncthreads();
  red[d] = kv * kv; __syncthreads();
  for (int st = 64; st > 0; st >>= 1) { if (d < st) red[d] += red[d + st]; __syncthreads(); }
  float kr = rsqrtf(red[0] * (1.f / Dh) + 1e-6f);
  __syncthreads();
  qsh[d] = qv * qr * qs[d];
  ksh[d] = kv * kr * ks[d];
  __syncthreads();
  int i = d >> 1, jj = d & 1;
  const float* per = pe + ((size_t)t * (Dh / 2) + i) * 4 + jj * 2;
  float qo = per[0] * qsh[2 * i] + per[1] * qsh[2 * i + 1];
  float ko = per[0] * ksh[2 * i] + per[1] * ksh[2 * i + 1];
  size_t o = ((size_t)hh * Lh + t) * Dh + d;
  qb[o] = (__bf16)qo; kb[o] = (__bf16)ko; vb[o] = (__bf16)vv;
}

// ---------------------------------------------------------------- V: [NH][L][D] -> [NH][D][L] (tiled)
__global__ void k_vtrans(const __bf16* __restrict__ v, __bf16* __restrict__ vt) {
  __shared__ __bf16 tile[32][33];
  int d0 = blockIdx.x * 32, t0 = blockIdx.y * 32, hh = blockIdx.z;
  int tx = threadIdx.x & 31, ty = threadIdx.x >> 5;
  const __bf16* vs = v + (size_t)hh * Lh * Dh;
  __bf16* vd = vt + (size_t)hh * Dh * Lh;
#pragma unroll
  for (int p = 0; p < 4; ++p)
    tile[ty + p * 8][tx] = vs[(size_t)(t0 + ty + p * 8) * Dh + d0 + tx];
  __syncthreads();
#pragma unroll
  for (int p = 0; p < 4; ++p)
    vd[(size_t)(d0 + ty + p * 8) * Lh + t0 + tx] = tile[tx][ty + p * 8];
}

// ---------------------------------------------------------------- GELU(mlp) -> act[:, 3072:]
__global__ void k_gelu(const float* __restrict__ h, __bf16* __restrict__ act) {
  for (size_t i = (size_t)blockIdx.x * 256 + threadIdx.x; i < (size_t)Lh * MLPh;
       i += (size_t)gridDim.x * 256) {
    size_t t = i / MLPh, c = i % MLPh;
    float xv = h[t * N1h + 3 * Hh + c];
    float g = 0.5f * xv * (1.f + tanhf(0.7978845608f * (xv + 0.044715f * xv * xv * xv)));
    act[t * K2h + Hh + c] = (__bf16)g;
  }
}

// ---------------------------------------------------------------- flash attention: one wave per 16 query rows
__device__ __forceinline__ int pcf(int c) {  // K-chunk permutation for A-fragment layout
  return (c & 7) | ((c & 8) << 1) | ((c & 16) >> 1);
}

__global__ __launch_bounds__(256) void k_attn(
    const __bf16* __restrict__ qb, const __bf16* __restrict__ kb,
    const __bf16* __restrict__ vt, __bf16* __restrict__ act) {
  __shared__ __bf16 pbuf[8][16][40];  // per-wave P transpose buffer (permuted cols)
  int tid = threadIdx.x;
  int wid = tid >> 5, lane = tid & 31;
  int half = lane >> 4, lp = lane & 15;
  int gw = blockIdx.x * 8 + wid;
  int hh = gw >> 7;
  int q0 = (gw & 127) << 4;
  const __bf16* qh  = qb + (size_t)hh * Lh * Dh;
  const __bf16* kh  = kb + (size_t)hh * Lh * Dh;
  const __bf16* vth = vt + (size_t)hh * Dh * Lh;
  int pcl0 = pcf(lp), pcl1 = pcf(16 + lp);

  v16bf qf[4];
  const __bf16* qrow = qh + (size_t)(q0 + lp) * Dh;
#pragma unroll
  for (int s = 0; s < 4; ++s) {
    v8bf lo = *(const v8bf*)(qrow + s * 32 + half * 8);
    v8bf hi = *(const v8bf*)(qrow + s * 32 + 16 + half * 8);
    qf[s] = CAT16(lo, hi);
  }
  float mi[8], li[8];
  v8f o[8] = {};
  for (int v = 0; v < 8; ++v) { mi[v] = -1e30f; li[v] = 0.f; }
  const float scl = 0.08838834764831845f;  // 1/sqrt(128)

  for (int kt = 0; kt < Lh; kt += 32) {
    if (kt + 32 < Lh) __builtin_prefetch(kh + (size_t)(kt + 32 + lp) * Dh, 0, 0);
    v8f sf[2] = {};
#pragma unroll
    for (int nn = 0; nn < 2; ++nn)
#pragma unroll
      for (int s = 0; s < 4; ++s) {
        const __bf16* kp = &kh[(size_t)(kt + nn * 16 + lp) * Dh + s * 32 + half * 8];
        v8bf lo = *(const v8bf*)kp;
        v8bf hi = *(const v8bf*)(kp + 16);
        v16bf kf = CAT16(lo, hi);
        sf[nn] = WMMA_BF16(qf[s], kf, sf[nn]);
      }
    float pr0[8], pr1[8];
#pragma unroll
    for (int v = 0; v < 8; ++v) {
      float s0 = sf[0][v] * scl, s1 = sf[1][v] * scl;
      float mx = fmaxf(s0, s1);
      for (int off = 1; off < 16; off <<= 1) mx = fmaxf(mx, __shfl_xor(mx, off, 32));
      float mn = fmaxf(mi[v], mx);
      float cf = __expf(mi[v] - mn);
      float p0 = __expf(s0 - mn), p1 = __expf(s1 - mn);
      float rs = p0 + p1;
      for (int off = 1; off < 16; off <<= 1) rs += __shfl_xor(rs, off, 32);
      li[v] = li[v] * cf + rs;
      mi[v] = mn;
#pragma unroll
      for (int j = 0; j < 8; ++j) o[j][v] *= cf;
      pr0[v] = p0; pr1[v] = p1;
    }
    // transpose P (C-layout) -> A-layout via per-wave LDS (in-wave DS ordering)
#pragma unroll
    for (int v = 0; v < 8; ++v) {
      pbuf[wid][v + 8 * half][pcl0] = (__bf16)pr0[v];
      pbuf[wid][v + 8 * half][pcl1] = (__bf16)pr1[v];
    }
    v8bf plo = *(const v8bf*)&pbuf[wid][lp][half * 16];
    v8bf phi = *(const v8bf*)&pbuf[wid][lp][half * 16 + 8];
    v16bf pf = CAT16(plo, phi);
#pragma unroll
    for (int j = 0; j < 8; ++j) {
      const __bf16* vp = &vth[(size_t)(j * 16 + lp) * Lh + kt + half * 8];
      v8bf lo = *(const v8bf*)vp;
      v8bf hi = *(const v8bf*)(vp + 16);
      v16bf vf = CAT16(lo, hi);
      o[j] = WMMA_BF16(pf, vf, o[j]);
    }
  }
  for (int j = 0; j < 8; ++j)
    for (int v = 0; v < 8; ++v) {
      float val = o[j][v] / li[v];
      act[(size_t)(q0 + v + 8 * half) * K2h + hh * Dh + j * 16 + lp] = (__bf16)val;
    }
}

// ---------------------------------------------------------------- out = x + gate * y
__global__ void k_residual(const float* __restrict__ x, const float* __restrict__ mod,
                           const float* __restrict__ y, float* __restrict__ out) {
  for (size_t i = (size_t)blockIdx.x * 256 + threadIdx.x; i < (size_t)Lh * Hh;
       i += (size_t)gridDim.x * 256) {
    int c = (int)(i % Hh);
    out[i] = x[i] + mod[2 * Hh + c] * y[i];
  }
}

// ================================================================ host
extern "C" void kernel_launch(void* const* d_in, const int* in_sizes, int n_in,
                              void* d_out, int out_size, void* d_ws, size_t ws_size,
                              hipStream_t stream) {
  const float* x        = (const float*)d_in[0];
  const float* vec      = (const float*)d_in[1];
  const float* pe       = (const float*)d_in[2];
  const float* mod_w    = (const float*)d_in[3];
  const float* mod_b    = (const float*)d_in[4];
  const float* ln_scale = (const float*)d_in[5];
  const float* ln_bias  = (const float*)d_in[6];
  const float* w1       = (const float*)d_in[7];
  const float* b1       = (const float*)d_in[8];
  const float* q_scale  = (const float*)d_in[9];
  const float* k_scale  = (const float*)d_in[10];
  const float* w2       = (const float*)d_in[11];
  const float* b2       = (const float*)d_in[12];

  char* ws = (char*)d_ws;
  size_t off = 0;
  auto carve = [&](size_t bytes) -> void* {
    void* p = ws + off;
    off = (off + bytes + 255) & ~(size_t)255;
    return p;
  };
  __bf16* w1t  = (__bf16*)carve((size_t)N1h * Hh * 2);   // [N1][H]
  __bf16* w2t  = (__bf16*)carve((size_t)Hh * K2h * 2);   // [H][K2]
  __bf16* xmod = (__bf16*)carve((size_t)Lh * Hh * 2);
  float*  hbuf = (float*)carve((size_t)Lh * N1h * 4);
  __bf16* qbuf = (__bf16*)carve((size_t)NHh * Lh * Dh * 2);
  __bf16* kbuf = (__bf16*)carve((size_t)NHh * Lh * Dh * 2);
  __bf16* vbuf = (__bf16*)carve((size_t)NHh * Lh * Dh * 2);
  __bf16* vtb  = (__bf16*)carve((size_t)NHh * Dh * Lh * 2);
  __bf16* act  = (__bf16*)carve((size_t)Lh * K2h * 2);
  float*  ybuf = (float*)carve((size_t)Lh * Hh * 4);
  float*  modb = (float*)carve((size_t)3 * Hh * 4);

  k_castT<<<dim3(Hh / 32, N1h / 32), 256, 0, stream>>>(w1, w1t, Hh, N1h);
  k_castT<<<dim3(K2h / 32, Hh / 32), 256, 0, stream>>>(w2, w2t, K2h, Hh);
  k_mod<<<(3 * Hh + 255) / 256, 256, 0, stream>>>(vec, mod_w, mod_b, modb);
  k_ln_mod<<<Lh, 256, 0, stream>>>(x, modb, ln_scale, ln_bias, xmod);
  k_gemm<<<dim3(N1h / 128, Lh / 128), 256, 0, stream>>>(xmod, w1t, b1, hbuf, Lh, N1h, Hh);
  k_qkrope<<<Lh * NHh, 128, 0, stream>>>(hbuf, pe, q_scale, k_scale, qbuf, kbuf, vbuf);
  k_vtrans<<<dim3(Dh / 32, Lh / 32, NHh), 256, 0, stream>>>(vbuf, vtb);
  k_gelu<<<4096, 256, 0, stream>>>(hbuf, act);
  k_attn<<<(Lh / 16) * NHh / 8, 256, 0, stream>>>(qbuf, kbuf, vtb, act);
  k_gemm<<<dim3(Hh / 128, Lh / 128), 256, 0, stream>>>(act, w2t, b2, ybuf, Lh, Hh, K2h);
  k_residual<<<4096, 256, 0, stream>>>(x, modb, ybuf, (float*)d_out);
}
